// GIN_2688649527604
// MI455X (gfx1250) — matrix operand order repaired and checked by reading, not compiled, not run on hardware
//
#include <hip/hip_runtime.h>

typedef __attribute__((ext_vector_type(16))) _Float16 v16h;
typedef __attribute__((ext_vector_type(8)))  float    v8f;

#define FEAT 128

// ---------------------------------------------------------------------------
// Edge scatter-add: one 32-lane "virtual wave slot" per edge, float4 per lane.
// agg[dst] += x[src]  (128 floats per edge). L2-resident atomics on MI455X.
// ---------------------------------------------------------------------------
__global__ void gin_scatter_add(const float* __restrict__ x,
                                const int* __restrict__ edge_index,
                                float* __restrict__ agg, int E) {
  long long gid = (long long)blockIdx.x * blockDim.x + threadIdx.x;
  int e = (int)(gid >> 5);
  if (e >= E) return;
  int lane = (int)(gid & 31);
  int s = edge_index[e];        // row 0 of (2,E)
  int d = edge_index[E + e];    // row 1 of (2,E)
  const float4 v = *(const float4*)(x + (size_t)s * FEAT + lane * 4);
  float* p = agg + (size_t)d * FEAT + lane * 4;
  atomicAdd(p + 0, v.x);
  atomicAdd(p + 1, v.y);
  atomicAdd(p + 2, v.z);
  atomicAdd(p + 3, v.w);
}

// ---------------------------------------------------------------------------
// Fused GIN MLP GEMM:  Out = act( [(1+eps)*X + AGG] @ W + Bias )
// If AGG==nullptr, A = X (plain linear).
// Block = 128 threads (4 wave32), 64 rows per block, all Nn columns.
// A staged fp32->f16 in LDS; W staged fp32->f16 transposed [n][k] in LDS.
// WMMA f32_16x16x32_f16, K-loop of K/32, N-loop of Nn/16 tiles.
// ---------------------------------------------------------------------------
__global__ void gin_mlp_gemm(const float* __restrict__ X,
                             const float* __restrict__ AGG,
                             const float* __restrict__ epsPtr,
                             const float* __restrict__ W,    // [K][Nn] row-major
                             const float* __restrict__ Bias, // [Nn]
                             float* __restrict__ Out,        // [M][Nn]
                             int M, int K, int Nn, int doRelu) {
  __shared__ _Float16 sA[64 * 128];    // 16 KB
  __shared__ _Float16 sW[128 * 128];   // 32 KB (transposed: [n][k])
  __shared__ float    sB[128];

  const int tid = threadIdx.x;
  const int rb  = blockIdx.x * 64;
  const float scale = (AGG != nullptr) ? (1.0f + *epsPtr) : 1.0f;

  // Stage combined A tile (coalesced float4 loads along k).
  for (int idx = tid * 4; idx < 64 * K; idx += (int)blockDim.x * 4) {
    int r = idx / K, k = idx % K;
    int m = rb + r;
    float4 xv = make_float4(0.f, 0.f, 0.f, 0.f);
    if (m < M) {
      xv = *(const float4*)(X + (size_t)m * K + k);
      if (AGG != nullptr) {
        float4 av = *(const float4*)(AGG + (size_t)m * K + k);
        xv.x = scale * xv.x + av.x;
        xv.y = scale * xv.y + av.y;
        xv.z = scale * xv.z + av.z;
        xv.w = scale * xv.w + av.w;
      }
    }
    _Float16* pa = &sA[r * K + k];
    pa[0] = (_Float16)xv.x; pa[1] = (_Float16)xv.y;
    pa[2] = (_Float16)xv.z; pa[3] = (_Float16)xv.w;
  }

  // Stage W transposed into LDS: sW[n*K + k] = W[k*Nn + n], fp32 -> f16.
  for (int idx = tid * 4; idx < K * Nn; idx += (int)blockDim.x * 4) {
    int k = idx / Nn, n = idx % Nn;  // Nn multiple of 4 -> n aligned to 4
    float4 wv = *(const float4*)(W + (size_t)k * Nn + n);
    sW[(n + 0) * K + k] = (_Float16)wv.x;
    sW[(n + 1) * K + k] = (_Float16)wv.y;
    sW[(n + 2) * K + k] = (_Float16)wv.z;
    sW[(n + 3) * K + k] = (_Float16)wv.w;
  }
  if (tid < Nn) sB[tid] = Bias[tid];
  __syncthreads();

  const int wave   = tid >> 5;
  const int lane   = tid & 31;
  const int hiHalf = (lane >> 4) & 1;        // lanes 16-31
  const int rowA   = wave * 16 + (lane & 15);
  const int kgrpA  = hiHalf * 8;             // A: lanes 16-31 hold K+8
  const int kgrpB  = hiHalf * 8;             // B: lanes 16-31 hold K+8
  const int ncol   = lane & 15;

  const int nTiles = Nn >> 4;
  const int kTiles = K >> 5;

  for (int nt = 0; nt < nTiles; ++nt) {
    v8f acc = {};
    const int n = nt * 16 + ncol;
    for (int kt = 0; kt < kTiles; ++kt) {
      const int ktb = kt * 32;
      union { v16h h; unsigned int u[8]; } a, b;
#pragma unroll
      for (int i = 0; i < 8; ++i) {
        // A 16x32 f16 layout: VGPR 0-3 -> K = kgrpA + {0..7} pairs,
        //                     VGPR 4-7 -> K = 16 + kgrpA + pairs.
        int kk = ((i < 4) ? (2 * i) : (16 + 2 * (i - 4))) + kgrpA;
        a.u[i] = *(const unsigned int*)&sA[rowA * K + ktb + kk];
        // B 32x16 f16 layout: same K mapping as A (kgrpB == kgrpA) at column n.
        b.u[i] = *(const unsigned int*)&sW[n * K + ktb + kk];
      }
      acc = __builtin_amdgcn_wmma_f32_16x16x32_f16(
          false, a.h, false, b.h, (short)0, acc, false, false);
    }
    const float bias = sB[n];
#pragma unroll
    for (int r = 0; r < 8; ++r) {
      int m = rb + wave * 16 + r + hiHalf * 8;  // C/D: VGPR r -> M=r(+8)
      if (m < M) {
        float v = acc[r] + bias;
        if (doRelu) v = fmaxf(v, 0.0f);
        Out[(size_t)m * Nn + n] = v;
      }
    }
  }
}

// ---------------------------------------------------------------------------
// Root gather: out[i,:] = h[roots[i],:]  (64 wide)
// ---------------------------------------------------------------------------
__global__ void gin_gather_roots(const float* __restrict__ h,
                                 const int* __restrict__ roots,
                                 float* __restrict__ out, int OUTF) {
  int i = blockIdx.x;
  int f = threadIdx.x;
  out[(size_t)i * OUTF + f] = h[(size_t)roots[i] * OUTF + f];
}

extern "C" void kernel_launch(void* const* d_in, const int* in_sizes, int n_in,
                              void* d_out, int out_size, void* d_ws, size_t ws_size,
                              hipStream_t stream) {
  const float* x     = (const float*)d_in[0];
  const int*   ei    = (const int*)d_in[1];
  const int*   roots = (const int*)d_in[2];
  const float* eps0  = (const float*)d_in[3];
  const float* w0a   = (const float*)d_in[4];
  const float* b0a   = (const float*)d_in[5];
  const float* w0b   = (const float*)d_in[6];
  const float* b0b   = (const float*)d_in[7];
  const float* eps1  = (const float*)d_in[8];
  const float* w1a   = (const float*)d_in[9];
  const float* b1a   = (const float*)d_in[10];
  const float* w1b   = (const float*)d_in[11];
  const float* b1b   = (const float*)d_in[12];
  const float* eps2  = (const float*)d_in[13];
  const float* w2a   = (const float*)d_in[14];
  const float* b2a   = (const float*)d_in[15];
  const float* w2b   = (const float*)d_in[16];
  const float* b2b   = (const float*)d_in[17];

  const int M = in_sizes[0] / FEAT;       // 100000 nodes
  const int E = in_sizes[1] / 2;          // 1.6M edges
  const int NR = in_sizes[2];             // 1024 roots

  const size_t per = (size_t)M * FEAT * sizeof(float);   // 51.2 MB
  char* ws = (char*)d_ws;
  float* agg = (float*)(ws + 0 * per);
  float* mid = (float*)(ws + 1 * per);
  float* hA  = (float*)(ws + 2 * per);
  float* hB  = (float*)(ws + 3 * per);

  dim3 sblk(256);
  dim3 sgrid((unsigned)(((long long)E * 32 + 255) / 256));
  dim3 gblk(128);
  dim3 ggrid((unsigned)((M + 63) / 64));

  // ---- Layer 0: h0 = relu( mlp0( (1+eps0)x + agg(x) ) ) ----
  hipMemsetAsync(agg, 0, per, stream);
  gin_scatter_add<<<sgrid, sblk, 0, stream>>>(x, ei, agg, E);
  gin_mlp_gemm<<<ggrid, gblk, 0, stream>>>(x, agg, eps0, w0a, b0a, mid, M, 128, 128, 1);
  gin_mlp_gemm<<<ggrid, gblk, 0, stream>>>(mid, nullptr, nullptr, w0b, b0b, hA, M, 128, 128, 1);

  // ---- Layer 1 ----
  hipMemsetAsync(agg, 0, per, stream);
  gin_scatter_add<<<sgrid, sblk, 0, stream>>>(hA, ei, agg, E);
  gin_mlp_gemm<<<ggrid, gblk, 0, stream>>>(hA, agg, eps1, w1a, b1a, mid, M, 128, 128, 1);
  gin_mlp_gemm<<<ggrid, gblk, 0, stream>>>(mid, nullptr, nullptr, w1b, b1b, hB, M, 128, 128, 1);

  // ---- Layer 2 (128 -> 64 -> 64, no outer relu) ----
  hipMemsetAsync(agg, 0, per, stream);
  gin_scatter_add<<<sgrid, sblk, 0, stream>>>(hB, ei, agg, E);
  gin_mlp_gemm<<<ggrid, gblk, 0, stream>>>(hB, agg, eps2, w2a, b2a, mid, M, 128, 64, 1);
  gin_mlp_gemm<<<ggrid, gblk, 0, stream>>>(mid, nullptr, nullptr, w2b, b2b, hA, M, 64, 64, 0);

  // ---- Root gather ----
  gin_gather_roots<<<dim3((unsigned)NR), dim3(64), 0, stream>>>(hA, roots, (float*)d_out, 64);
}
